// MixtureOfExperts_30210799960341
// MI455X (gfx1250) — matrix operand order repaired and checked by reading, not compiled, not run on hardware
//
#include <hip/hip_runtime.h>
#include <hip/hip_bf16.h>

// Problem sizes (fixed by the reference)
#define NTOK 8192
#define DIM  512
#define HID  2048
#define OUT  512
#define NEXP 8

typedef __attribute__((ext_vector_type(16))) __bf16 v16bf;
typedef __attribute__((ext_vector_type(8)))  __bf16 v8bf;
typedef __attribute__((ext_vector_type(8)))  float  v8f;
typedef __attribute__((ext_vector_type(4)))  unsigned int v4u;
typedef __attribute__((ext_vector_type(8)))  int          v8i;
typedef __attribute__((ext_vector_type(4)))  int          v4i;

// ---------------- workspace layout (bytes) ----------------
#define OFF_XBF      ((size_t)0)                       // N*D bf16        = 8 MiB
#define OFF_W1T      ((size_t)8388608)                 // E*H*D bf16      = 16 MiB  [e][h][d]
#define OFF_W2T      ((size_t)25165824)                // E*O*H bf16      = 16 MiB  [e][o][h]
#define OFF_H        ((size_t)41943040)                // N*H bf16        = 32 MiB  (compact slots)
#define OFF_LISTS    ((size_t)75497472)                // E*N int
#define OFF_COMPACT  ((size_t)75759616)                // N int
#define OFF_CNT      ((size_t)75792384)                // 8 counters + 9 offsets (int)

#if __has_builtin(__builtin_amdgcn_tensor_load_to_lds) && \
    __has_builtin(__builtin_amdgcn_s_wait_tensorcnt)
#define MOE_HAVE_TDM 1
#endif

__device__ __forceinline__ v16bf pack16(v8bf lo, v8bf hi) {
  v16bf r;
#pragma unroll
  for (int i = 0; i < 8; ++i) { r[i] = lo[i]; r[i + 8] = hi[i]; }
  return r;
}

// Low 32 bits of a generic pointer into the LDS aperture == LDS byte offset.
__device__ __forceinline__ unsigned lds_addr(const void* p) {
  return (unsigned)(uintptr_t)p;
}

#ifdef MOE_HAVE_TDM
// 2D tile DMA: tile_w (2-byte elements) x tile_h rows, row stride in elements.
// D# layout per CDNA5 ISA.
// This toolchain exposes the 6-arg builtin: (v4u g0, v8i g1, v4i g2, v4i g3, v8i g4, i32 cpol).
__device__ __forceinline__ void tdm_load_tile_2d(unsigned lds_byte_addr, const void* gptr,
                                                 unsigned tile_w, unsigned tile_h,
                                                 unsigned row_stride_elems) {
  unsigned long long ga = (unsigned long long)(uintptr_t)gptr;
  v4u g0;
  g0[0] = 1u;                                                   // count=1, user descriptor
  g0[1] = lds_byte_addr;                                        // lds_addr
  g0[2] = (unsigned)(ga & 0xFFFFFFFFu);                         // global_addr lo
  g0[3] = (unsigned)((ga >> 32) & 0x01FFFFFFu) | 0x80000000u;   // global_addr hi | type=2
  const unsigned td0 = 1u << 20, td1 = 1u << 20;                // huge tensor dims: no clipping
  v8i g1;
  g1[0] = (int)(1u << 16);                                      // data_size=1 (2 bytes)
  g1[1] = (int)((td0 & 0xFFFFu) << 16);                         // barrier_addr=0 | dim0.lo16
  g1[2] = (int)(((td0 >> 16) & 0xFFFFu) | ((td1 & 0xFFFFu) << 16));
  g1[3] = (int)(((td1 >> 16) & 0xFFFFu) | (tile_w << 16));      // dim1.hi16 | tile_dim0
  g1[4] = (int)(tile_h & 0xFFFFu);                              // tile_dim1 | tile_dim2=0
  g1[5] = (int)row_stride_elems;                                // tensor_dim0_stride lo32
  g1[6] = 0;
  g1[7] = 0;
  v4i gz  = {0, 0, 0, 0};
  v8i gz8 = {0, 0, 0, 0, 0, 0, 0, 0};
  __builtin_amdgcn_tensor_load_to_lds(g0, g1, gz, gz, gz8, 0);
}
#endif

// ---------------- tiny utility kernels ----------------
__global__ void moe_zero_counters(int* counters) {
  if (threadIdx.x < 8) counters[threadIdx.x] = 0;
}

__global__ void moe_cvt_x(const float* __restrict__ x, __bf16* __restrict__ xbf, int n) {
  int i = (blockIdx.x * blockDim.x + threadIdx.x) * 4;
  if (i < n) {
    float4 v = *(const float4*)(x + i);
    xbf[i + 0] = (__bf16)v.x;
    xbf[i + 1] = (__bf16)v.y;
    xbf[i + 2] = (__bf16)v.z;
    xbf[i + 3] = (__bf16)v.w;
  }
}

// src: [rows][cols] f32 per expert -> dst: [cols][rows] bf16 per expert
__global__ void moe_transpose_cvt(const float* __restrict__ src, __bf16* __restrict__ dst,
                                  int rows, int cols) {
  __shared__ float tile[32][33];
  const float* s = src + (size_t)blockIdx.z * rows * cols;
  __bf16*      d = dst + (size_t)blockIdx.z * rows * cols;
  int c0 = blockIdx.x * 32, r0 = blockIdx.y * 32;
  int tx = threadIdx.x & 31, ty = threadIdx.x >> 5;   // 32 x 8
#pragma unroll
  for (int i = 0; i < 32; i += 8)
    tile[ty + i][tx] = s[(size_t)(r0 + ty + i) * cols + (c0 + tx)];
  __syncthreads();
#pragma unroll
  for (int i = 0; i < 32; i += 8)
    d[(size_t)(c0 + ty + i) * rows + (r0 + tx)] = (__bf16)tile[tx][ty + i];
}

// ---------------- router: logits -> softmax -> argmax, build expert lists ----------------
__global__ void moe_router(const float* __restrict__ x, const float* __restrict__ Wr,
                           const float* __restrict__ br, float* __restrict__ probs_out,
                           int* __restrict__ lists, int* __restrict__ counters) {
  int wave = threadIdx.x >> 5, lane = threadIdx.x & 31;
  int token = blockIdx.x * 8 + wave;          // 8 waves per block, one token per wave
  const float* xrow = x + (size_t)token * DIM;
  float acc[NEXP];
#pragma unroll
  for (int e = 0; e < NEXP; ++e) acc[e] = 0.f;
  for (int d = lane; d < DIM; d += 32) {
    float xv = xrow[d];
    const float4 w0 = *(const float4*)(Wr + d * NEXP);
    const float4 w1 = *(const float4*)(Wr + d * NEXP + 4);
    acc[0] += xv * w0.x; acc[1] += xv * w0.y; acc[2] += xv * w0.z; acc[3] += xv * w0.w;
    acc[4] += xv * w1.x; acc[5] += xv * w1.y; acc[6] += xv * w1.z; acc[7] += xv * w1.w;
  }
#pragma unroll
  for (int m = 16; m >= 1; m >>= 1)
#pragma unroll
    for (int e = 0; e < NEXP; ++e) acc[e] += __shfl_xor(acc[e], m, 32);

#pragma unroll
  for (int e = 0; e < NEXP; ++e) acc[e] += br[e];

  float mx = acc[0];
#pragma unroll
  for (int e = 1; e < NEXP; ++e) mx = fmaxf(mx, acc[e]);
  float p[NEXP], sum = 0.f;
#pragma unroll
  for (int e = 0; e < NEXP; ++e) { p[e] = __expf(acc[e] - mx); sum += p[e]; }
  float inv = 1.f / sum;
#pragma unroll
  for (int e = 0; e < NEXP; ++e) p[e] *= inv;

  if (lane < NEXP) probs_out[(size_t)token * NEXP + lane] = p[lane];

  if (lane == 0) {
    int amax = 0; float best = p[0];
#pragma unroll
    for (int e = 1; e < NEXP; ++e) if (p[e] > best) { best = p[e]; amax = e; }
    int pos = atomicAdd(&counters[amax], 1);
    lists[amax * NTOK + pos] = token;
  }
}

__global__ void moe_finalize(const int* __restrict__ counters, int* __restrict__ offsets,
                             float* __restrict__ counts_out) {
  if (threadIdx.x == 0) {
    int o = 0;
#pragma unroll
    for (int e = 0; e < NEXP; ++e) {
      offsets[e] = o;
      counts_out[e] = (float)counters[e];
      o += counters[e];
    }
    offsets[NEXP] = o;
  }
}

__global__ void moe_compact(const int* __restrict__ lists, const int* __restrict__ counters,
                            const int* __restrict__ offsets, int* __restrict__ compact) {
  int e = blockIdx.x;
  int c = counters[e], o = offsets[e];
  for (int i = threadIdx.x; i < c; i += blockDim.x)
    compact[o + i] = lists[e * NTOK + i];
}

// ---------------- layer 1: h = relu(x_g @ W1[e] + b1[e]) -> bf16, compact slots ----------------
// grid: (HID/64, NTOK/128, NEXP), block 256 (8 waves). Tile M=128 tokens x N=64 hidden cols.
// A (gathered tokens) staged to LDS by all threads; B (weights) staged by TDM, double-buffered.
__global__ void __launch_bounds__(256) moe_mlp1(
    const __bf16* __restrict__ xbf, const __bf16* __restrict__ w1t,
    const float* __restrict__ b1, const int* __restrict__ compact,
    const int* __restrict__ counters, const int* __restrict__ offsets,
    __bf16* __restrict__ hbf) {
  __shared__ __align__(16) __bf16 Abuf[2][128][32];
  __shared__ __align__(16) __bf16 Bbuf[2][64][32];

  const int e   = blockIdx.z;
  const int cnt = counters[e];
  const int m0  = blockIdx.y * 128;
  if (m0 >= cnt) return;
  const int n0   = blockIdx.x * 64;
  const int tid  = threadIdx.x;
  const int wave = tid >> 5, lane = tid & 31;
  const int off  = offsets[e];

  // staging assignment: A = 128 rows x 32 elems; 256 threads x 32B
  const int srow = tid >> 1;
  const int sseg = (tid & 1) * 16;
  int sgm = m0 + srow; if (sgm >= cnt) sgm = cnt - 1;        // clamp gather
  const __bf16* a_src = xbf + (size_t)compact[off + sgm] * DIM + sseg;

  const __bf16* b_base = w1t + ((size_t)e * HID + n0) * DIM; // TDM tile origin
#ifndef MOE_HAVE_TDM
  const int brow_t = tid >> 2;
  const int bseg   = (tid & 3) * 8;
  const __bf16* b_src = w1t + ((size_t)e * HID + (n0 + brow_t)) * DIM + bseg;
#endif

  // prologue: stage k=0 into buffer 0
  *(uint4*)&Abuf[0][srow][sseg]     = *(const uint4*)(a_src);
  *(uint4*)&Abuf[0][srow][sseg + 8] = *(const uint4*)(a_src + 8);
#ifdef MOE_HAVE_TDM
  if (wave == 0) tdm_load_tile_2d(lds_addr(&Bbuf[0][0][0]), b_base, 32, 64, DIM);
#else
  *(uint4*)&Bbuf[0][brow_t][bseg]   = *(const uint4*)(b_src);
#endif

  const int mw = (wave >> 2) * 64, nw = (wave & 3) * 16;
  const int lrow = lane & 15, lhalf = lane >> 4;
  const int ka = lhalf * 16;               // A frag: K {ka..ka+15}
  const int kb = lhalf * 16;               // B frag: K {kb..kb+15}
  v8f acc[4] = {};

  for (int it = 0; it < DIM / 32; ++it) {
    const int cur = it & 1, nxt = cur ^ 1;
    const int knext = (it + 1) * 32;
#ifdef MOE_HAVE_TDM
    if (wave == 0) __builtin_amdgcn_s_wait_tensorcnt(0);     // cur B tile landed
#endif
    __syncthreads();                                         // cur buffers visible
    if (knext < DIM) {
      __builtin_prefetch(a_src + knext + 32, 0, 1);
      *(uint4*)&Abuf[nxt][srow][sseg]     = *(const uint4*)(a_src + knext);
      *(uint4*)&Abuf[nxt][srow][sseg + 8] = *(const uint4*)(a_src + knext + 8);
#ifdef MOE_HAVE_TDM
      if (wave == 0) tdm_load_tile_2d(lds_addr(&Bbuf[nxt][0][0]), b_base + knext, 32, 64, DIM);
#else
      *(uint4*)&Bbuf[nxt][brow_t][bseg]   = *(const uint4*)(b_src + knext);
#endif
    }
    v16bf B = pack16(*(const v8bf*)&Bbuf[cur][nw + lrow][kb],
                     *(const v8bf*)&Bbuf[cur][nw + lrow][kb + 8]);
#pragma unroll
    for (int t = 0; t < 4; ++t) {
      v16bf A = pack16(*(const v8bf*)&Abuf[cur][mw + t * 16 + lrow][ka],
                       *(const v8bf*)&Abuf[cur][mw + t * 16 + lrow][ka + 8]);
      acc[t] = __builtin_amdgcn_wmma_f32_16x16x32_bf16(false, A, false, B,
                                                       (short)0, acc[t], false, false);
    }
    __syncthreads();                                         // done reading cur
  }

  const int ncol = n0 + nw + lrow;
  const float bias = b1[e * HID + ncol];
#pragma unroll
  for (int t = 0; t < 4; ++t)
#pragma unroll
    for (int v = 0; v < 8; ++v) {
      int gm = m0 + mw + t * 16 + (lhalf ? 8 + v : v);       // C: VGPR v -> M=v / M=v+8
      if (gm < cnt) {
        float val = acc[t][v] + bias;
        hbf[(size_t)(off + gm) * HID + ncol] = (__bf16)(val > 0.f ? val : 0.f);
      }
    }
}

// ---------------- layer 2: out[token] = h_slot @ W2[e] + b2[e], scatter to d_out ----------------
// grid: (OUT/64, NTOK/128, NEXP), block 256.
__global__ void __launch_bounds__(256) moe_mlp2(
    const __bf16* __restrict__ hbf, const __bf16* __restrict__ w2t,
    const float* __restrict__ b2, const int* __restrict__ compact,
    const int* __restrict__ counters, const int* __restrict__ offsets,
    float* __restrict__ out) {
  __shared__ __align__(16) __bf16 Abuf[2][128][32];
  __shared__ __align__(16) __bf16 Bbuf[2][64][32];

  const int e   = blockIdx.z;
  const int cnt = counters[e];
  const int m0  = blockIdx.y * 128;
  if (m0 >= cnt) return;
  const int n0   = blockIdx.x * 64;
  const int tid  = threadIdx.x;
  const int wave = tid >> 5, lane = tid & 31;
  const int off  = offsets[e];

  const int srow = tid >> 1;
  const int sseg = (tid & 1) * 16;
  int sgm = m0 + srow; if (sgm >= cnt) sgm = cnt - 1;
  const __bf16* a_src = hbf + (size_t)(off + sgm) * HID + sseg;   // compact slots

  const __bf16* b_base = w2t + ((size_t)e * OUT + n0) * HID;
#ifndef MOE_HAVE_TDM
  const int brow_t = tid >> 2;
  const int bseg   = (tid & 3) * 8;
  const __bf16* b_src = w2t + ((size_t)e * OUT + (n0 + brow_t)) * HID + bseg;
#endif

  *(uint4*)&Abuf[0][srow][sseg]     = *(const uint4*)(a_src);
  *(uint4*)&Abuf[0][srow][sseg + 8] = *(const uint4*)(a_src + 8);
#ifdef MOE_HAVE_TDM
  if (wave == 0) tdm_load_tile_2d(lds_addr(&Bbuf[0][0][0]), b_base, 32, 64, HID);
#else
  *(uint4*)&Bbuf[0][brow_t][bseg]   = *(const uint4*)(b_src);
#endif

  const int mw = (wave >> 2) * 64, nw = (wave & 3) * 16;
  const int lrow = lane & 15, lhalf = lane >> 4;
  const int ka = lhalf * 16;
  const int kb = lhalf * 16;
  v8f acc[4] = {};

  for (int it = 0; it < HID / 32; ++it) {
    const int cur = it & 1, nxt = cur ^ 1;
    const int knext = (it + 1) * 32;
#ifdef MOE_HAVE_TDM
    if (wave == 0) __builtin_amdgcn_s_wait_tensorcnt(0);
#endif
    __syncthreads();
    if (knext < HID) {
      __builtin_prefetch(a_src + knext + 32, 0, 1);
      *(uint4*)&Abuf[nxt][srow][sseg]     = *(const uint4*)(a_src + knext);
      *(uint4*)&Abuf[nxt][srow][sseg + 8] = *(const uint4*)(a_src + knext + 8);
#ifdef MOE_HAVE_TDM
      if (wave == 0) tdm_load_tile_2d(lds_addr(&Bbuf[nxt][0][0]), b_base + knext, 32, 64, HID);
#else
      *(uint4*)&Bbuf[nxt][brow_t][bseg]   = *(const uint4*)(b_src + knext);
#endif
    }
    v16bf B = pack16(*(const v8bf*)&Bbuf[cur][nw + lrow][kb],
                     *(const v8bf*)&Bbuf[cur][nw + lrow][kb + 8]);
#pragma unroll
    for (int t = 0; t < 4; ++t) {
      v16bf A = pack16(*(const v8bf*)&Abuf[cur][mw + t * 16 + lrow][ka],
                       *(const v8bf*)&Abuf[cur][mw + t * 16 + lrow][ka + 8]);
      acc[t] = __builtin_amdgcn_wmma_f32_16x16x32_bf16(false, A, false, B,
                                                       (short)0, acc[t], false, false);
    }
    __syncthreads();
  }

  const int ncol = n0 + nw + lrow;
  const float bias = b2[e * OUT + ncol];
#pragma unroll
  for (int t = 0; t < 4; ++t)
#pragma unroll
    for (int v = 0; v < 8; ++v) {
      int gm = m0 + mw + t * 16 + (lhalf ? 8 + v : v);
      if (gm < cnt) {
        int token = compact[off + gm];
        out[(size_t)token * OUT + ncol] = acc[t][v] + bias;   // forward scale == 1
      }
    }
}

// ---------------- launcher ----------------
extern "C" void kernel_launch(void* const* d_in, const int* in_sizes, int n_in,
                              void* d_out, int out_size, void* d_ws, size_t ws_size,
                              hipStream_t stream) {
  (void)in_sizes; (void)n_in; (void)out_size; (void)ws_size;
  const float* x  = (const float*)d_in[0];
  const float* Wr = (const float*)d_in[1];
  const float* br = (const float*)d_in[2];
  const float* W1 = (const float*)d_in[3];
  const float* b1 = (const float*)d_in[4];
  const float* W2 = (const float*)d_in[5];
  const float* b2 = (const float*)d_in[6];

  float* out_logits = (float*)d_out;                              // [N, O]
  float* out_probs  = out_logits + (size_t)NTOK * OUT;            // [N, E]
  float* out_counts = out_probs + (size_t)NTOK * NEXP;            // [E]

  char* ws = (char*)d_ws;
  __bf16* xbf      = (__bf16*)(ws + OFF_XBF);
  __bf16* w1t      = (__bf16*)(ws + OFF_W1T);
  __bf16* w2t      = (__bf16*)(ws + OFF_W2T);
  __bf16* hbf      = (__bf16*)(ws + OFF_H);
  int*    lists    = (int*)(ws + OFF_LISTS);
  int*    compact  = (int*)(ws + OFF_COMPACT);
  int*    counters = (int*)(ws + OFF_CNT);
  int*    offsets  = counters + 8;

  moe_zero_counters<<<1, 32, 0, stream>>>(counters);

  moe_cvt_x<<<(NTOK * DIM / 4 + 255) / 256, 256, 0, stream>>>(x, xbf, NTOK * DIM);

  // W1 [E, D, H] -> w1t [E, H, D]
  moe_transpose_cvt<<<dim3(HID / 32, DIM / 32, NEXP), 256, 0, stream>>>(W1, w1t, DIM, HID);
  // W2 [E, H, O] -> w2t [E, O, H]
  moe_transpose_cvt<<<dim3(OUT / 32, HID / 32, NEXP), 256, 0, stream>>>(W2, w2t, HID, OUT);

  moe_router<<<NTOK / 8, 256, 0, stream>>>(x, Wr, br, out_probs, lists, counters);
  moe_finalize<<<1, 32, 0, stream>>>(counters, offsets, out_counts);
  moe_compact<<<NEXP, 256, 0, stream>>>(lists, counters, offsets, compact);

  moe_mlp1<<<dim3(HID / 64, NTOK / 128, NEXP), 256, 0, stream>>>(
      xbf, w1t, b1, compact, counters, offsets, hbf);
  moe_mlp2<<<dim3(OUT / 64, NTOK / 128, NEXP), 256, 0, stream>>>(
      hbf, w2t, b2, compact, counters, offsets, out_logits);
}
